// MultiHeadLatentAttentionWithoutRoPE_60378650247250
// MI455X (gfx1250) — matrix operand order repaired and checked by reading, not compiled, not run on hardware
//
#include <hip/hip_runtime.h>
#include <hip/hip_bf16.h>

// MLA without RoPE, bf16 WMMA pipeline for gfx1250 (wave32).
// B=2, S=2048, D_MODEL=1024, N_HEADS=16, D_HEAD=64, D_KV=256.
// Weights + V are pre-packed into WMMA B-fragment-native swizzled bf16 tiles
// (32K x 16N tile = 32 lanes x 16 contiguous bf16) so every fragment load is
// 1-2 wide vector loads per lane with no in-loop f32->bf16 conversion.
// WGP-scope global_prefetch_b8 (inline asm, scope 0) issued ahead in hot loops.

typedef __bf16 bf16_t;
typedef __attribute__((ext_vector_type(16))) __bf16 v16bf;
typedef __attribute__((ext_vector_type(8)))  __bf16 v8bf;
typedef __attribute__((ext_vector_type(8)))  float  v8f;

#define DEV static __device__ __forceinline__

// WGP-scope prefetch: scope field 0 pulls into all cache levels on miss.
// (__builtin_prefetch only maps to SE/SYS scope, which skip the WGP cache.)
DEV void prefetch_wgp(const void* p) {
  asm volatile("global_prefetch_b8 %0, off" :: "v"(p));
}

// ---------- WMMA fragment helpers (wave32, 16x16x32 bf16) ----------
// A (16x32, MxK): lanes 0-15 = rows, elems 0-15: K=kb..kb+15 (kb = 16*half)
DEV v16bf load_a_bf16(const bf16_t* A, int lda, int lane) {
  const int row = lane & 15;
  const int kb  = (lane >> 4) << 4;
  const bf16_t* p = A + (size_t)row * lda + kb;
  const v8bf lo = *(const v8bf*)(p);
  const v8bf hi = *(const v8bf*)(p + 8);
  return __builtin_shufflevector(lo, hi, 0, 1, 2, 3, 4, 5, 6, 7,
                                 8, 9, 10, 11, 12, 13, 14, 15);
}

// B (32x16, KxN) from row-major [N x K] transposed source (Q*K^T path):
// lane = (half -> K base 0/16) + col; 16 contiguous elements per lane.
DEV v16bf load_bT_bf16(const bf16_t* Mrow, int ldm, int lane) {
  const int col = lane & 15;
  const int kb  = (lane >> 4) << 4;
  return *(const v16bf*)(Mrow + (size_t)col * ldm + kb);
}

// B from packed swizzled layout: tile (kt,nt) = 512 bf16, lane-major.
DEV v16bf load_b_packed(const bf16_t* Wsw, int ntiles, int kc, int n0, int lane) {
  const size_t tile = (size_t)(kc >> 5) * ntiles + (n0 >> 4);
  return *(const v16bf*)(Wsw + tile * 512 + lane * 16);
}

DEV v8f wmma_bf16(v16bf a, v16bf b, v8f c) {
  return __builtin_amdgcn_wmma_f32_16x16x32_bf16(false, a, false, b, (short)0, c,
                                                 false, false);
}

// Store a 16x16 f32 D fragment into the swizzled bf16 layout (used for V).
DEV void store_sw(bf16_t* out, int ntiles, int mt, int nc, int lane, v8f acc) {
  const int roff = (lane >> 4) * 8;
  const int col  = lane & 15;
  const size_t tile = (size_t)(mt >> 5) * ntiles + (nc >> 4);
  bf16_t* p = out + tile * 512 + (size_t)((((mt >> 4) & 1) * 16 + col) * 16) + roff;
#pragma unroll
  for (int r = 0; r < 8; ++r) p[r] = (bf16_t)acc[r];
}

// ---------- Pack kernels ----------
__global__ __launch_bounds__(256)
void pack_x_bf16(const float* __restrict__ x, bf16_t* __restrict__ out) {
  const size_t base = ((size_t)blockIdx.x * 256 + threadIdx.x) * 16;
#pragma unroll
  for (int i = 0; i < 16; ++i) out[base + i] = (bf16_t)x[base + i];
}

// W [K x N] f32 row-major -> swizzled bf16 tiles. One thread = one lane-chunk.
__global__ __launch_bounds__(256)
void pack_w_sw(const float* __restrict__ W, bf16_t* __restrict__ out, int N) {
  const int g      = blockIdx.x * 256 + threadIdx.x;
  const int laneid = g & 31;
  const int tile   = g >> 5;
  const int ntiles = N >> 4;
  const int kt = tile / ntiles;
  const int nt = tile - kt * ntiles;
  const int half = laneid >> 4;
  const int colx = laneid & 15;
  const float* src = W + (size_t)(kt * 32 + half * 16) * N + nt * 16 + colx;
  bf16_t* dst = out + (size_t)g * 16;
#pragma unroll
  for (int i = 0; i < 16; ++i) dst[i] = (bf16_t)src[(size_t)i * N];
}

// ---------- Kernel 1: c = rmsnorm(x @ W_d, ln_w), bf16 out ----------
__global__ __launch_bounds__(256)
void down_proj_rmsnorm(const bf16_t* __restrict__ x, const bf16_t* __restrict__ Wsw,
                       const float* __restrict__ ln_w, bf16_t* __restrict__ out) {
  __shared__ float tile[16 * 256];
  __shared__ float scale[16];
  const int tid  = threadIdx.x;
  const int lane = tid & 31;
  const int wave = tid >> 5;
  const int m0   = blockIdx.x * 16;
  const int n0   = wave * 32;

  v8f acc0 = {}, acc1 = {};
  const bf16_t* Arow = x + (size_t)m0 * 1024;
  for (int kc = 0; kc < 1024; kc += 32) {
    if (kc + 64 < 1024) {
      prefetch_wgp(Arow + (size_t)(lane & 15) * 1024 + kc + 64);
      prefetch_wgp(Wsw + ((size_t)((kc + 64) >> 5) * 16 + (n0 >> 4)) * 512 + lane * 16);
    }
    v16bf a  = load_a_bf16(Arow + kc, 1024, lane);
    v16bf b0 = load_b_packed(Wsw, 16, kc, n0, lane);
    v16bf b1 = load_b_packed(Wsw, 16, kc, n0 + 16, lane);
    acc0 = wmma_bf16(a, b0, acc0);
    acc1 = wmma_bf16(a, b1, acc1);
  }
  const int roff = (lane >> 4) * 8;
  const int col  = lane & 15;
#pragma unroll
  for (int r = 0; r < 8; ++r) {
    tile[(roff + r) * 256 + n0 + col]      = acc0[r];
    tile[(roff + r) * 256 + n0 + 16 + col] = acc1[r];
  }
  __syncthreads();
  if (tid < 16) {
    float s = 0.f;
    for (int c = 0; c < 256; ++c) { float v = tile[tid * 256 + c]; s += v * v; }
    scale[tid] = rsqrtf(s * (1.0f / 256.0f) + 1e-5f);
  }
  __syncthreads();
  const int row = tid >> 4;
  const int cb  = (tid & 15) * 16;
  const float sc = scale[row];
  bf16_t* orow = out + (size_t)(m0 + row) * 256;
#pragma unroll
  for (int e = 0; e < 16; ++e) {
    const int c = cb + e;
    orow[c] = (bf16_t)(tile[row * 256 + c] * sc * ln_w[c]);
  }
}

// ---------- Kernel 2: up projection c[4096x256] @ W[256x1024] ----------
// 2x2 tiles per wave (32M x 32N), block = 64M x 128N.
__global__ __launch_bounds__(256)
void up_proj(const bf16_t* __restrict__ C, const bf16_t* __restrict__ Wsw,
             bf16_t* __restrict__ out, int vswizzle) {
  const int lane = threadIdx.x & 31;
  const int wave = threadIdx.x >> 5;
  const int m0 = blockIdx.x * 64 + (wave >> 2) * 32;
  const int n0 = blockIdx.y * 128 + (wave & 3) * 32;

  v8f a00 = {}, a01 = {}, a10 = {}, a11 = {};
  const bf16_t* A0 = C + (size_t)m0 * 256;
  const bf16_t* A1 = C + (size_t)(m0 + 16) * 256;
  for (int kc = 0; kc < 256; kc += 32) {
    if (kc + 64 < 256) {
      prefetch_wgp(A0 + (size_t)(lane & 15) * 256 + kc + 64);
      prefetch_wgp(A1 + (size_t)(lane & 15) * 256 + kc + 64);
      prefetch_wgp(Wsw + ((size_t)((kc + 64) >> 5) * 64 + (n0 >> 4)) * 512 + lane * 16);
    }
    const v16bf fa0 = load_a_bf16(A0 + kc, 256, lane);
    const v16bf fa1 = load_a_bf16(A1 + kc, 256, lane);
    const v16bf fb0 = load_b_packed(Wsw, 64, kc, n0, lane);
    const v16bf fb1 = load_b_packed(Wsw, 64, kc, n0 + 16, lane);
    a00 = wmma_bf16(fa0, fb0, a00);
    a01 = wmma_bf16(fa0, fb1, a01);
    a10 = wmma_bf16(fa1, fb0, a10);
    a11 = wmma_bf16(fa1, fb1, a11);
  }
  const int roff = (lane >> 4) * 8;
  const int col  = lane & 15;
  if (!vswizzle) {
#pragma unroll
    for (int r = 0; r < 8; ++r) {
      const size_t r0 = (size_t)(m0 + roff + r) * 1024;
      const size_t r1 = (size_t)(m0 + 16 + roff + r) * 1024;
      out[r0 + n0 + col]      = (bf16_t)a00[r];
      out[r0 + n0 + 16 + col] = (bf16_t)a01[r];
      out[r1 + n0 + col]      = (bf16_t)a10[r];
      out[r1 + n0 + 16 + col] = (bf16_t)a11[r];
    }
  } else {
    store_sw(out, 64, m0,      n0,      lane, a00);
    store_sw(out, 64, m0,      n0 + 16, lane, a01);
    store_sw(out, 64, m0 + 16, n0,      lane, a10);
    store_sw(out, 64, m0 + 16, n0 + 16, lane, a11);
  }
}

// ---------- Kernel 3: causal flash attention, one 16-row block per wave ----------
// mblk = blockIdx.x + wave*16 balances causal work across workgroups.
__global__ __launch_bounds__(256)
void mla_attention(const bf16_t* __restrict__ Q, const bf16_t* __restrict__ Kmat,
                   const bf16_t* __restrict__ Vsw, const int* __restrict__ amask,
                   bf16_t* __restrict__ O) {
  __shared__ __align__(32) bf16_t pbuf[8][16 * 32];
  const int tid  = threadIdx.x;
  const int lane = tid & 31;
  const int wave = tid >> 5;
  const int h    = blockIdx.y;
  const int b    = blockIdx.z;
  const int m0   = (blockIdx.x + wave * 16) * 16;

  const size_t rowbase = (size_t)b * 2048;
  const bf16_t* qp = Q    + (rowbase + m0) * 1024 + h * 64;
  const bf16_t* kp = Kmat + rowbase * 1024 + h * 64;
  const int*   amp = amask + b * 2048;

  const v16bf aq0 = load_a_bf16(qp,      1024, lane);   // d = 0..31
  const v16bf aq1 = load_a_bf16(qp + 32, 1024, lane);   // d = 32..63

  v8f o0 = {}, o1 = {}, o2 = {}, o3 = {};
  float m_i[8], l_i[8];
  const float NEG_INF = -__builtin_inff();
#pragma unroll
  for (int r = 0; r < 8; ++r) { m_i[r] = NEG_INF; l_i[r] = 0.f; }

  const int roff  = (lane >> 4) * 8;
  const int col   = lane & 15;
  const int kbmax = (m0 + 15) >> 5;
  bf16_t* pt = pbuf[wave];

  for (int kb = 0; kb <= kbmax; ++kb) {
    const int key0 = kb * 32;
    if (kb < kbmax) {  // prefetch next chunk's K rows and first V tile
      prefetch_wgp(kp + (size_t)(key0 + 32 + lane) * 1024);
      prefetch_wgp(Vsw + ((size_t)((rowbase + key0 + 32) >> 5) * 64 + h * 4) * 512 +
                   lane * 16);
    }
    // S = Q * K^T (two 16-key tiles, each with two 32-deep d chunks)
    v8f s0 = {}, s1 = {};
    s0 = wmma_bf16(aq0, load_bT_bf16(kp + (size_t)key0 * 1024,             1024, lane), s0);
    s0 = wmma_bf16(aq1, load_bT_bf16(kp + (size_t)key0 * 1024 + 32,        1024, lane), s0);
    s1 = wmma_bf16(aq0, load_bT_bf16(kp + (size_t)(key0 + 16) * 1024,      1024, lane), s1);
    s1 = wmma_bf16(aq1, load_bT_bf16(kp + (size_t)(key0 + 16) * 1024 + 32, 1024, lane), s1);

    const int key_a = key0 + col;
    const int key_b = key0 + 16 + col;
    const bool ma = (amp[key_a] != 0);
    const bool mb = (amp[key_b] != 0);

#pragma unroll
    for (int r = 0; r < 8; ++r) {
      const int row = m0 + roff + r;
      float f0 = (ma && key_a <= row) ? s0[r] * 0.125f : NEG_INF;
      float f1 = (mb && key_b <= row) ? s1[r] * 0.125f : NEG_INF;
      // row reduction over the 16 lanes of this half-wave
      float rmx = fmaxf(f0, f1);
      rmx = fmaxf(rmx, __shfl_xor(rmx, 1));
      rmx = fmaxf(rmx, __shfl_xor(rmx, 2));
      rmx = fmaxf(rmx, __shfl_xor(rmx, 4));
      rmx = fmaxf(rmx, __shfl_xor(rmx, 8));
      const float mnew  = fmaxf(m_i[r], rmx);
      const float alpha = __expf(m_i[r] - mnew);
      const float p0 = __expf(f0 - mnew);
      const float p1 = __expf(f1 - mnew);
      s0[r] = p0; s1[r] = p1;
      float rs = p0 + p1;
      rs += __shfl_xor(rs, 1);
      rs += __shfl_xor(rs, 2);
      rs += __shfl_xor(rs, 4);
      rs += __shfl_xor(rs, 8);
      l_i[r] = l_i[r] * alpha + rs;
      m_i[r] = mnew;
      o0[r] *= alpha; o1[r] *= alpha; o2[r] *= alpha; o3[r] *= alpha;
    }

    // D-fragment -> A-fragment relayout through per-wave LDS tile (16x32 bf16)
#pragma unroll
    for (int r = 0; r < 8; ++r) {
      pt[(roff + r) * 32 + col]      = (bf16_t)s0[r];
      pt[(roff + r) * 32 + 16 + col] = (bf16_t)s1[r];
    }
    const v16bf ap = load_a_bf16(pt, 32, lane);

    // O += P * V, V in swizzled layout: one 32B vector load per fragment.
    const size_t vt = (size_t)((rowbase + key0) >> 5) * 64 + h * 4;
    const bf16_t* vb = Vsw + vt * 512 + lane * 16;
    o0 = wmma_bf16(ap, *(const v16bf*)(vb),        o0);
    o1 = wmma_bf16(ap, *(const v16bf*)(vb + 512),  o1);
    o2 = wmma_bf16(ap, *(const v16bf*)(vb + 1024), o2);
    o3 = wmma_bf16(ap, *(const v16bf*)(vb + 1536), o3);
  }

  bf16_t* op = O + (rowbase + m0) * 1024 + h * 64;
#pragma unroll
  for (int r = 0; r < 8; ++r) {
    const float inv = 1.0f / l_i[r];
    bf16_t* orow = op + (size_t)(roff + r) * 1024;
    orow[col]      = (bf16_t)(o0[r] * inv);
    orow[16 + col] = (bf16_t)(o1[r] * inv);
    orow[32 + col] = (bf16_t)(o2[r] * inv);
    orow[48 + col] = (bf16_t)(o3[r] * inv);
  }
}

// ---------- Kernel 4: out = attn_out @ fc_w + fc_b (f32 out), 2x2 tiles ----------
__global__ __launch_bounds__(256)
void final_fc(const bf16_t* __restrict__ A, const bf16_t* __restrict__ Wsw,
              const float* __restrict__ bias, float* __restrict__ out) {
  const int lane = threadIdx.x & 31;
  const int wave = threadIdx.x >> 5;
  const int m0 = blockIdx.x * 64 + (wave >> 2) * 32;
  const int n0 = blockIdx.y * 128 + (wave & 3) * 32;

  v8f a00 = {}, a01 = {}, a10 = {}, a11 = {};
  const bf16_t* A0 = A + (size_t)m0 * 1024;
  const bf16_t* A1 = A + (size_t)(m0 + 16) * 1024;
  for (int kc = 0; kc < 1024; kc += 32) {
    if (kc + 64 < 1024) {
      prefetch_wgp(A0 + (size_t)(lane & 15) * 1024 + kc + 64);
      prefetch_wgp(A1 + (size_t)(lane & 15) * 1024 + kc + 64);
      prefetch_wgp(Wsw + ((size_t)((kc + 64) >> 5) * 64 + (n0 >> 4)) * 512 + lane * 16);
    }
    const v16bf fa0 = load_a_bf16(A0 + kc, 1024, lane);
    const v16bf fa1 = load_a_bf16(A1 + kc, 1024, lane);
    const v16bf fb0 = load_b_packed(Wsw, 64, kc, n0, lane);
    const v16bf fb1 = load_b_packed(Wsw, 64, kc, n0 + 16, lane);
    a00 = wmma_bf16(fa0, fb0, a00);
    a01 = wmma_bf16(fa0, fb1, a01);
    a10 = wmma_bf16(fa1, fb0, a10);
    a11 = wmma_bf16(fa1, fb1, a11);
  }
  const int roff = (lane >> 4) * 8;
  const int col  = lane & 15;
  const float b0 = bias[n0 + col];
  const float b1 = bias[n0 + 16 + col];
#pragma unroll
  for (int r = 0; r < 8; ++r) {
    const size_t r0 = (size_t)(m0 + roff + r) * 1024;
    const size_t r1 = (size_t)(m0 + 16 + roff + r) * 1024;
    out[r0 + n0 + col]      = a00[r] + b0;
    out[r0 + n0 + 16 + col] = a01[r] + b1;
    out[r1 + n0 + col]      = a10[r] + b0;
    out[r1 + n0 + 16 + col] = a11[r] + b1;
  }
}

// ---------- Host launch ----------
extern "C" void kernel_launch(void* const* d_in, const int* in_sizes, int n_in,
                              void* d_out, int out_size, void* d_ws, size_t ws_size,
                              hipStream_t stream) {
  const float* x     = (const float*)d_in[0];
  const int*   amask = (const int*)d_in[1];
  const float* W_dq  = (const float*)d_in[2];
  const float* W_uq  = (const float*)d_in[3];
  const float* W_dkv = (const float*)d_in[4];
  const float* W_uk  = (const float*)d_in[5];
  const float* W_uv  = (const float*)d_in[6];
  const float* q_ln  = (const float*)d_in[7];
  const float* kv_ln = (const float*)d_in[8];
  const float* fc_w  = (const float*)d_in[9];
  const float* fc_b  = (const float*)d_in[10];
  float* out = (float*)d_out;

  char* ws = (char*)d_ws;
  const size_t MB = 1u << 20;
  bf16_t* xb    = (bf16_t*)(ws);                    // 4096x1024 bf16 = 8MB
  bf16_t* c_q   = (bf16_t*)(ws + 8 * MB);           // 2MB
  bf16_t* c_kv  = (bf16_t*)(ws + 10 * MB);          // 2MB
  bf16_t* q     = (bf16_t*)(ws + 12 * MB);          // 8MB
  bf16_t* k     = (bf16_t*)(ws + 20 * MB);          // 8MB
  bf16_t* v_sw  = (bf16_t*)(ws + 28 * MB);          // 8MB (swizzled)
  bf16_t* ao    = (bf16_t*)(ws + 36 * MB);          // 8MB
  bf16_t* dq_sw = (bf16_t*)(ws + 44 * MB);          // 512KB
  bf16_t* dk_sw = (bf16_t*)(ws + 44 * MB + 512 * 1024);
  bf16_t* uq_sw = (bf16_t*)(ws + 45 * MB);
  bf16_t* uk_sw = (bf16_t*)(ws + 45 * MB + 512 * 1024);
  bf16_t* uv_sw = (bf16_t*)(ws + 46 * MB);
  bf16_t* fc_sw = (bf16_t*)(ws + 47 * MB);          // 2MB; total 49MB

  const dim3 blk(256);
  pack_x_bf16<<<dim3(1024), blk, 0, stream>>>(x, xb);
  pack_w_sw<<<dim3(64),  blk, 0, stream>>>(W_dq,  dq_sw, 256);
  pack_w_sw<<<dim3(64),  blk, 0, stream>>>(W_dkv, dk_sw, 256);
  pack_w_sw<<<dim3(64),  blk, 0, stream>>>(W_uq,  uq_sw, 1024);
  pack_w_sw<<<dim3(64),  blk, 0, stream>>>(W_uk,  uk_sw, 1024);
  pack_w_sw<<<dim3(64),  blk, 0, stream>>>(W_uv,  uv_sw, 1024);
  pack_w_sw<<<dim3(256), blk, 0, stream>>>(fc_w,  fc_sw, 1024);

  down_proj_rmsnorm<<<dim3(256), blk, 0, stream>>>(xb, dq_sw, q_ln,  c_q);
  down_proj_rmsnorm<<<dim3(256), blk, 0, stream>>>(xb, dk_sw, kv_ln, c_kv);
  up_proj<<<dim3(64, 8), blk, 0, stream>>>(c_q,  uq_sw, q,    0);
  up_proj<<<dim3(64, 8), blk, 0, stream>>>(c_kv, uk_sw, k,    0);
  up_proj<<<dim3(64, 8), blk, 0, stream>>>(c_kv, uv_sw, v_sw, 1);
  mla_attention<<<dim3(16, 16, 2), blk, 0, stream>>>(q, k, v_sw, amask, ao);
  final_fc<<<dim3(64, 8), blk, 0, stream>>>(ao, fc_sw, fc_b, out);
}
